// Codebook_40072044871897
// MI455X (gfx1250) — matrix operand order repaired and checked by reading, not compiled, not run on hardware
//
#include <hip/hip_runtime.h>
#include <hip/hip_bf16.h>

// Problem dims (compile-time constants from the reference)
#define BATCH 16
#define TT    4096
#define DI    512     // D_IN
#define DE    256     // D_EMB
#define NC    1024    // N_CODES
#define BT    (BATCH*TT)   // 65536 rows

typedef __bf16 bf16;
typedef __attribute__((ext_vector_type(8)))  __bf16 v8bf;
typedef __attribute__((ext_vector_type(16))) __bf16 v16bf;
typedef __attribute__((ext_vector_type(8)))  float  v8f;

union BF16x16 { v16bf v; v8bf h[2]; };

// ---- CDNA5 WMMA fragment loaders (wave32, 16x16x32 bf16) ----
// A matrix 16x32 (MxK): lane L holds row m=L&15; kh=L>>4 selects the two
// 8-element contiguous K-chunks: [k+kh*8, +8) and [k+16+kh*8, +8).
__device__ __forceinline__ v16bf load_a_frag(const bf16* __restrict__ row, int k, int kh) {
  BF16x16 u;
  u.h[0] = *(const v8bf*)(row + k + kh * 8);
  u.h[1] = *(const v8bf*)(row + k + 16 + kh * 8);
  return u.v;
}
// B matrix 32x16 (KxN): lane L holds column n=L&15; kh=L>>4 selects the two
// 8-element contiguous K-chunks: [k+kh*8, +8) and [k+16+kh*8, +8).
__device__ __forceinline__ v16bf load_b_frag(const bf16* __restrict__ rowN, int k, int kh) {
  BF16x16 u;
  u.h[0] = *(const v8bf*)(rowN + k + kh * 8);
  u.h[1] = *(const v8bf*)(rowN + k + 16 + kh * 8);
  return u.v;
}

__device__ __forceinline__ v8f wmma_bf16(v16bf a, v16bf b, v8f c) {
  // (neg_a, A, neg_b, B, c_mod, C, reuse_a, reuse_b)
  return __builtin_amdgcn_wmma_f32_16x16x32_bf16(false, a, false, b, (short)0, c, false, false);
}

// Scheduling fence: forces the tied fragments to be loaded (defined) before
// this point and kept live in distinct VGPRs, preserving the batched
// load-clause -> WMMA-chain shape (and letting the scheduler hoist the next
// k-step's loads under the current WMMAs). Kept at 4 fragments so the
// pipelined live set stays within the register file (no spills).
#define PIN4(a0, a1, a2, a3) \
  asm volatile("" : "+v"(a0), "+v"(a1), "+v"(a2), "+v"(a3))

// ------------- f32 -> bf16 conversion, 8 elems/iter (b128 in/out) -------------
__global__ void k_cvt8(const float* __restrict__ in, bf16* __restrict__ out, int n) {
  int i = blockIdx.x * blockDim.x + threadIdx.x;
  const int stride = gridDim.x * blockDim.x;
  const int n8 = n >> 3;
  for (; i < n8; i += stride) {
    const float4* p = (const float4*)in + 2 * (size_t)i;
    float4 x = p[0], y = p[1];
    v8bf o = { (bf16)x.x, (bf16)x.y, (bf16)x.z, (bf16)x.w,
               (bf16)y.x, (bf16)y.y, (bf16)y.z, (bf16)y.w };
    *((v8bf*)out + i) = o;
  }
}

// ------------- per-code squared norms: one wave per code, coalesced -------------
__global__ void k_norms(const float* __restrict__ emb, float* __restrict__ norms) {
  const int lane = threadIdx.x & 31;
  const int c = blockIdx.x * (blockDim.x >> 5) + (threadIdx.x >> 5);
  const float4* r = (const float4*)(emb + (size_t)c * DE) + lane * 2;
  float4 x = r[0], y = r[1];
  float s = x.x * x.x + x.y * x.y + x.z * x.z + x.w * x.w
          + y.x * y.x + y.y * y.y + y.z * y.z + y.w * y.w;
  #pragma unroll
  for (int off = 1; off < 32; off <<= 1) s += __shfl_xor(s, off, 32);
  if (lane == 0) norms[c] = s;
}

// ---------------- GEMM1: z_e = z @ W_down^T  (K=512) ----------------
// One wave per 16x64 tile. Per k-step: one batched 10-load clause (A + 4 B
// fragments, B pinned live), then 4 WMMAs; the scheduler pipelines the next
// k-step's loads under the current WMMAs within register budget.
__global__ void k_proj_down(const bf16* __restrict__ zb, const bf16* __restrict__ wdb,
                            float* __restrict__ ze_f, bf16* __restrict__ ze_b) {
  const int lane = threadIdx.x & 31;
  const int wave = blockIdx.x * (blockDim.x >> 5) + (threadIdx.x >> 5);
  const int row0 = (wave >> 2) * 16;   // 4096 row tiles
  const int col0 = (wave & 3) * 64;    // 4 col groups of 64 (DE=256)
  const int mr = lane & 15;
  const int kh = lane >> 4;
  const bf16* arow = zb  + (size_t)(row0 + mr) * DI;
  const bf16* brow = wdb + (size_t)(col0 + mr) * DI;   // n == mr
  v8f acc[4];
  #pragma unroll
  for (int j = 0; j < 4; ++j) acc[j] = (v8f){};
  #pragma unroll
  for (int k = 0; k < DI; k += 32) {
    __builtin_prefetch(arow + k + 128, 0, 3);   // prefetch streaming z operand
    v16bf af = load_a_frag(arow, k, kh);
    v16bf bfr[4];
    #pragma unroll
    for (int j = 0; j < 4; ++j) bfr[j] = load_b_frag(brow + j * 16 * DI, k, kh);
    PIN4(bfr[0], bfr[1], bfr[2], bfr[3]);
    #pragma unroll
    for (int j = 0; j < 4; ++j) acc[j] = wmma_bf16(af, bfr[j], acc[j]);
  }
  // C/D layout: lane holds col n=lane&15, rows v + 8*kh
  #pragma unroll
  for (int v = 0; v < 8; ++v) {
    const int mm = row0 + v + 8 * kh;
    float* of = ze_f + (size_t)mm * DE + col0 + mr;
    bf16*  ob = ze_b + (size_t)mm * DE + col0 + mr;
    #pragma unroll
    for (int j = 0; j < 4; ++j) {
      of[j * 16] = acc[j][v];
      ob[j * 16] = (bf16)acc[j][v];
    }
  }
}

// ---------------- Quantize: argmin_n (||c_n||^2 - 2 z_e . c_n) ----------------
// A row (16x256 bf16 = 8 fragments = 64 VGPRs/lane) stays register-resident
// across all code tiles. Two code tiles run concurrently (independent acc
// chains); kk unrolled by 2 so each pinned 8-load clause feeds 4 interleaved
// WMMAs from the two chains.
__global__ void k_quantize(const bf16* __restrict__ ze_b, const bf16* __restrict__ emb_b,
                           const float* __restrict__ norms, int* __restrict__ code_i,
                           float* __restrict__ code_f) {
  const int lane = threadIdx.x & 31;
  const int wave = blockIdx.x * (blockDim.x >> 5) + (threadIdx.x >> 5);
  const int row0 = wave * 16;
  const int mr = lane & 15;
  const int kh = lane >> 4;
  const bf16* arow = ze_b + (size_t)(row0 + mr) * DE;
  v16bf afr[8];
  #pragma unroll
  for (int kk = 0; kk < 8; ++kk) afr[kk] = load_a_frag(arow, kk * 32, kh);

  float bestd[8]; int besti[8];
  #pragma unroll
  for (int v = 0; v < 8; ++v) { bestd[v] = 3.4e38f; besti[v] = 0; }

  for (int c0 = 0; c0 < NC; c0 += 32) {     // two 16-code tiles per iteration
    const bf16* br0 = emb_b + (size_t)(c0 + mr) * DE;
    const bf16* br1 = emb_b + (size_t)(c0 + 16 + mr) * DE;
    v8f acc0 = {}, acc1 = {};
    #pragma unroll
    for (int kk = 0; kk < 8; kk += 2) {
      v16bf bfr[4];
      bfr[0] = load_b_frag(br0, kk * 32, kh);
      bfr[1] = load_b_frag(br1, kk * 32, kh);
      bfr[2] = load_b_frag(br0, (kk + 1) * 32, kh);
      bfr[3] = load_b_frag(br1, (kk + 1) * 32, kh);
      PIN4(bfr[0], bfr[1], bfr[2], bfr[3]);
      acc0 = wmma_bf16(afr[kk], bfr[0], acc0);
      acc1 = wmma_bf16(afr[kk], bfr[1], acc1);
      acc0 = wmma_bf16(afr[kk + 1], bfr[2], acc0);
      acc1 = wmma_bf16(afr[kk + 1], bfr[3], acc1);
    }
    const float nrm0 = norms[c0 + mr];
    const float nrm1 = norms[c0 + 16 + mr];
    #pragma unroll
    for (int v = 0; v < 8; ++v) {
      float d0 = nrm0 - 2.0f * acc0[v];
      float d1 = nrm1 - 2.0f * acc1[v];
      if (d0 < bestd[v]) { bestd[v] = d0; besti[v] = c0 + mr; }        // lower idx first
      if (d1 < bestd[v]) { bestd[v] = d1; besti[v] = c0 + 16 + mr; }
    }
  }
  // Butterfly argmin across the 16 lanes holding the same row
  // (xor offsets 1,2,4,8 stay within each 16-lane half of the wave32).
  #pragma unroll
  for (int off = 1; off < 16; off <<= 1) {
    #pragma unroll
    for (int v = 0; v < 8; ++v) {
      float od = __shfl_xor(bestd[v], off, 32);
      int   oi = __shfl_xor(besti[v], off, 32);
      if (od < bestd[v] || (od == bestd[v] && oi < besti[v])) { bestd[v] = od; besti[v] = oi; }
    }
  }
  if (mr == 0) {
    #pragma unroll
    for (int v = 0; v < 8; ++v) {
      const int r = row0 + v + 8 * kh;
      code_i[r] = besti[v];
      code_f[r] = (float)besti[v];
    }
  }
}

// ---------------- Loss: deterministic two-stage reduction ----------------
__global__ void k_loss_partial(const float* __restrict__ ze, const float* __restrict__ emb,
                               const int* __restrict__ code, float* __restrict__ part) {
  __shared__ float red[256];
  const int row0 = blockIdx.x * 16;          // 16 rows, never straddles a batch
  float s = 0.f;
  for (int i = threadIdx.x; i < 16 * DE; i += 256) {
    const int r = row0 + (i >> 8);
    const int e = i & (DE - 1);
    float d = emb[(size_t)code[r] * DE + e] - ze[(size_t)r * DE + e];
    s += d * d;
  }
  red[threadIdx.x] = s; __syncthreads();
  for (int o = 128; o > 0; o >>= 1) {
    if (threadIdx.x < o) red[threadIdx.x] += red[threadIdx.x + o];
    __syncthreads();
  }
  if (threadIdx.x == 0) part[blockIdx.x] = red[0];
}

__global__ void k_loss_final(const float* __restrict__ part, float* __restrict__ loss) {
  __shared__ float red[256];
  const int b = blockIdx.x;                  // 256 partials per batch
  red[threadIdx.x] = part[b * 256 + threadIdx.x];
  __syncthreads();
  for (int o = 128; o > 0; o >>= 1) {
    if (threadIdx.x < o) red[threadIdx.x] += red[threadIdx.x + o];
    __syncthreads();
  }
  if (threadIdx.x == 0) {
    float v = red[0] * (1.0f / ((float)TT * (float)DE));
    loss[b] = v;        // commitment_loss
    loss[16 + b] = v;   // codebook_loss (same forward value)
  }
}

// ---------------- GEMM3: z_q_out = emb[code] @ W_up^T  (K=256, gather-A) ----------------
// A (gathered embedding row) fully register-resident; 16x64 tile; per
// k-step one pinned 8-load clause (4 B fragments) feeds 4 WMMAs.
__global__ void k_proj_up(const bf16* __restrict__ emb_b, const bf16* __restrict__ wub,
                          const int* __restrict__ code, float* __restrict__ outp) {
  const int lane = threadIdx.x & 31;
  const int wave = blockIdx.x * (blockDim.x >> 5) + (threadIdx.x >> 5);
  const int row0 = (wave >> 3) * 16;   // 4096 row tiles
  const int col0 = (wave & 7) * 64;    // 8 col groups of 64 (DI=512)
  const int mr = lane & 15;
  const int kh = lane >> 4;
  const bf16* arow = emb_b + (size_t)code[row0 + mr] * DE;   // gathered A row
  const bf16* brow = wub + (size_t)(col0 + mr) * DE;
  v16bf afr[8];
  #pragma unroll
  for (int kk = 0; kk < 8; ++kk) afr[kk] = load_a_frag(arow, kk * 32, kh);

  v8f acc[4];
  #pragma unroll
  for (int j = 0; j < 4; ++j) acc[j] = (v8f){};
  #pragma unroll
  for (int kk = 0; kk < 8; ++kk) {
    v16bf bfr[4];
    #pragma unroll
    for (int j = 0; j < 4; ++j) bfr[j] = load_b_frag(brow + j * 16 * DE, kk * 32, kh);
    PIN4(bfr[0], bfr[1], bfr[2], bfr[3]);
    #pragma unroll
    for (int j = 0; j < 4; ++j) acc[j] = wmma_bf16(afr[kk], bfr[j], acc[j]);
  }
  #pragma unroll
  for (int v = 0; v < 8; ++v) {
    const int mm = row0 + v + 8 * kh;
    float* o = outp + (size_t)mm * DI + col0 + mr;
    #pragma unroll
    for (int j = 0; j < 4; ++j) o[j * 16] = acc[j][v];
  }
}

extern "C" void kernel_launch(void* const* d_in, const int* in_sizes, int n_in,
                              void* d_out, int out_size, void* d_ws, size_t ws_size,
                              hipStream_t stream) {
  const float* z   = (const float*)d_in[0];   // [16,4096,512]
  const float* emb = (const float*)d_in[1];   // [1024,256]
  const float* wd  = (const float*)d_in[2];   // [256,512]
  const float* wu  = (const float*)d_in[3];   // [512,256]

  // Output layout (flat concat, return order)
  float* out    = (float*)d_out;
  float* o_zq   = out;                          // 16*4096*512
  float* o_loss = out + (size_t)BT * DI;        // 16 + 16
  float* o_code = o_loss + 32;                  // 16*4096 (as float)
  float* o_ze   = o_code + BT;                  // 16*4096*256

  // bf16 staging of z (64MB) and z_e (32MB) lives INSIDE the 128MB z_q_out
  // region; it is fully consumed before k_proj_up overwrites it.
  bf16* z_bf  = (bf16*)o_zq;
  bf16* ze_bf = (bf16*)(o_zq + (size_t)16 * 1024 * 1024);  // byte offset 64MB

  // Workspace (~1.3MB): norms | code ints | loss partials | bf16 weights
  char* w = (char*)d_ws;
  float* norms  = (float*)w;  w += NC * sizeof(float);
  int*   code_i = (int*)w;    w += (size_t)BT * sizeof(int);
  float* part   = (float*)w;  w += (BT / 16) * sizeof(float);
  bf16*  wd_bf  = (bf16*)w;   w += (size_t)DE * DI * sizeof(bf16);
  bf16*  emb_bf = (bf16*)w;   w += (size_t)NC * DE * sizeof(bf16);
  bf16*  wu_bf  = (bf16*)w;

  k_cvt8<<<2048, 256, 0, stream>>>(z,   z_bf,  BT * DI);
  k_cvt8<<<64,   256, 0, stream>>>(wd,  wd_bf, DE * DI);
  k_cvt8<<<128,  256, 0, stream>>>(emb, emb_bf, NC * DE);
  k_cvt8<<<64,   256, 0, stream>>>(wu,  wu_bf, DI * DE);
  k_norms<<<NC / 8, 256, 0, stream>>>(emb, norms);   // 8 waves/block, 1 code/wave

  // 16384 waves (8 per 256-thread block), 16x64 tiles
  k_proj_down<<<2048, 256, 0, stream>>>(z_bf, wd_bf, o_ze, ze_bf);
  // 4096 waves
  k_quantize<<<512, 256, 0, stream>>>(ze_bf, emb_bf, norms, code_i, o_code);
  k_loss_partial<<<BT / 16, 256, 0, stream>>>(o_ze, emb, code_i, part);
  k_loss_final<<<BATCH, 256, 0, stream>>>(part, o_loss);
  // 32768 waves, 16x64 tiles
  k_proj_up<<<4096, 256, 0, stream>>>(emb_bf, wu_bf, code_i, o_zq);
}
